// MultiHeadAttention_5987184410842
// MI455X (gfx1250) — compile-verified, matrix operand order checked by reading, not run on hardware
//
#include <hip/hip_runtime.h>
#include <hip/hip_bf16.h>

typedef __bf16 bf16;
typedef __attribute__((ext_vector_type(16))) __bf16 v16bf;
typedef __attribute__((ext_vector_type(4)))  __bf16 v4bf;
typedef __attribute__((ext_vector_type(8)))  float  v8f;
typedef __attribute__((ext_vector_type(4)))  unsigned int u32x4;
typedef __attribute__((ext_vector_type(8)))  int i32x8;
typedef __attribute__((ext_vector_type(4)))  int i32x4;

#if __has_builtin(__builtin_amdgcn_tensor_load_to_lds)
#define HAVE_TDM 1
#else
#define HAVE_TDM 0
#endif

#define WMMA_BF16(a, b, c) \
  __builtin_amdgcn_wmma_f32_16x16x32_bf16(false, (a), false, (b), (short)0, (c), false, false)

// ---------------------------------------------------------------------------
// TDM helpers
// ---------------------------------------------------------------------------
__device__ __forceinline__ unsigned lds_off(const void* p) {
  // Flat addresses into LDS carry the workgroup-relative offset in addr[31:0].
  return (unsigned)(uintptr_t)p;
}

#if HAVE_TDM
// 2D bf16 tile load: tile_w elems per row, tile_h rows, row stride in elems.
// Padding codes: pad_int: interval = 2^(c+1) DWORDs; pad_amt: amount = c+1 DWORDs.
__device__ __forceinline__ void tdm_load_2d(unsigned lds_addr, const bf16* gptr,
                                            unsigned tile_w, unsigned tile_h,
                                            unsigned long long stride_elems,
                                            unsigned pad_int, unsigned pad_amt) {
  unsigned long long ga = (unsigned long long)(uintptr_t)gptr;
  u32x4 g0;
  g0[0] = 1u;                                                   // count=1, no gather
  g0[1] = lds_addr;                                             // lds_addr [63:32]
  g0[2] = (unsigned)ga;                                         // global_addr [95:64]
  g0[3] = (unsigned)((ga >> 32) & 0x01ffffffu) | (2u << 30);    // addr[120:96], type=2
  i32x8 g1;
  g1[0] = (int)((1u << 16) | (1u << 20) | (pad_int << 22) | (pad_amt << 25));
  g1[1] = (int)((tile_w & 0xffffu) << 16);                      // tensor_dim0 lo16 [63:48]
  g1[2] = (int)(((tile_w >> 16) & 0xffffu) | ((tile_h & 0xffffu) << 16)); // dim0 hi, dim1 lo
  g1[3] = (int)(((tile_h >> 16) & 0xffffu) | ((tile_w & 0xffffu) << 16)); // dim1 hi, tile_dim0
  g1[4] = (int)(tile_h & 0xffffu);                              // tile_dim1; tile_dim2=0
  g1[5] = (int)(unsigned)(stride_elems & 0xffffffffu);          // dim0_stride lo32
  g1[6] = (int)(unsigned)((stride_elems >> 32) & 0xffffu);      // dim0_stride hi16
  g1[7] = 0;
  i32x4 z4 = {0, 0, 0, 0};
#if __clang_major__ >= 23
  i32x8 z8 = {0, 0, 0, 0, 0, 0, 0, 0};
  __builtin_amdgcn_tensor_load_to_lds(g0, g1, z4, z4, z8, 0);
#else
  __builtin_amdgcn_tensor_load_to_lds(g0, g1, z4, z4, 0);
#endif
}
#define TDM_WAIT(n) __builtin_amdgcn_s_wait_tensorcnt(n)
#define DS_FENCE() asm volatile("s_wait_dscnt 0x0" ::: "memory")
#endif

// ---------------------------------------------------------------------------
// WMMA fragment builders (ISA 7.12.2 layouts)
// ---------------------------------------------------------------------------
__device__ __forceinline__ int kmap(int e, int half) {
  return ((e >> 3) << 4) + (half << 3) + (((e >> 1) & 3) << 1) + (e & 1);
}

__device__ __forceinline__ v16bf load_a(const bf16* lds, int row0, int k0, int S) {
  int lane = threadIdx.x & 31;
  int half = lane >> 4, r = lane & 15;
  v16bf a;
#pragma unroll
  for (int e = 0; e < 16; ++e) a[e] = lds[(row0 + r) * S + k0 + kmap(e, half)];
  return a;
}

__device__ __forceinline__ v16bf load_b(const bf16* lds, int k0, int n0, int S) {
  int lane = threadIdx.x & 31;
  int half = lane >> 4, n = lane & 15;
  v16bf b;
#pragma unroll
  for (int e = 0; e < 16; ++e) b[e] = lds[(k0 + kmap(e, half)) * S + n0 + n];
  return b;
}

// Transposed gather: LDS tile stored [n][k]; B[k][n] = lds[(n0+n)*S + k0+k].
__device__ __forceinline__ v16bf load_bt(const bf16* lds, int k0, int n0, int S) {
  int lane = threadIdx.x & 31;
  int half = lane >> 4, n = lane & 15;
  v16bf b;
#pragma unroll
  for (int e = 0; e < 16; ++e) b[e] = lds[(n0 + n) * S + k0 + kmap(e, half)];
  return b;
}

// ---------------------------------------------------------------------------
// fp32 -> bf16 bulk convert
// ---------------------------------------------------------------------------
__global__ __launch_bounds__(256) void cvt_bf16_kernel(const float* __restrict__ src,
                                                       bf16* __restrict__ dst, int n) {
  int i = (blockIdx.x * 256 + threadIdx.x) * 4;
  if (i + 3 < n) {
    float4 v = *(const float4*)(src + i);
    v4bf o = {(bf16)v.x, (bf16)v.y, (bf16)v.z, (bf16)v.w};
    *(v4bf*)(dst + i) = o;
  }
}

// ---------------------------------------------------------------------------
// GEMM with bias: Out[M,N] = A[M,K]*W[K,N] + bias[N]; A,W bf16, bias f32.
// 128 threads (4 waves); block tile 64x64; wave tile 32x32; double-buffered TDM.
// ---------------------------------------------------------------------------
template <bool OUT_BF16>
__global__ __launch_bounds__(128) void gemm_bias_kernel(
    const bf16* __restrict__ A, const bf16* __restrict__ W,
    const float* __restrict__ bias, void* __restrict__ Out,
    int M, int N, int K) {
  constexpr int AS = 40;  // 32 + 4-DWORD pad
  constexpr int BS = 72;  // 64 + 4-DWORD pad
  __shared__ bf16 As[2][64 * AS];
  __shared__ bf16 Bs[2][32 * BS];

  const int m0 = blockIdx.x * 64;
  const int n0 = blockIdx.y * 64;
  const int tid = threadIdx.x;
  const int wave = tid >> 5;
  const int wm = (wave >> 1) * 32;
  const int wn = (wave & 1) * 32;
  const int lane = tid & 31;
  const int half = lane >> 4, nlo = lane & 15;

  v8f acc[2][2];
#pragma unroll
  for (int i = 0; i < 2; ++i)
#pragma unroll
    for (int j = 0; j < 2; ++j) acc[i][j] = v8f{};

  const int nsteps = K / 32;

#if HAVE_TDM
  if (tid < 32) {  // wave 0 drives the tensor DMA
    tdm_load_2d(lds_off(&As[0][0]), A + (size_t)m0 * K, 32, 64, (unsigned long long)K, 3, 3);
    tdm_load_2d(lds_off(&Bs[0][0]), W + n0, 64, 32, (unsigned long long)N, 4, 3);
  }
#endif

  for (int s = 0; s < nsteps; ++s) {
    const int cur = s & 1, nxt = cur ^ 1;
#if HAVE_TDM
    if (tid < 32) {
      if (s + 1 < nsteps) {
        const int kt = (s + 1) * 32;
        tdm_load_2d(lds_off(&As[nxt][0]), A + (size_t)m0 * K + kt, 32, 64,
                    (unsigned long long)K, 3, 3);
        tdm_load_2d(lds_off(&Bs[nxt][0]), W + (size_t)kt * N + n0, 64, 32,
                    (unsigned long long)N, 4, 3);
        TDM_WAIT(2);  // current pair landed; next pair may be in flight
      } else {
        TDM_WAIT(0);
      }
    }
    __syncthreads();
#else
    {  // cooperative fallback
      const int kt = s * 32;
#pragma unroll
      for (int i = 0; i < 16; ++i) {
        int lin = tid + i * 128;
        int r = lin >> 5, c = lin & 31;
        As[cur][r * AS + c] = A[(size_t)(m0 + r) * K + kt + c];
      }
#pragma unroll
      for (int i = 0; i < 16; ++i) {
        int lin = tid + i * 128;
        int r = lin >> 6, c = lin & 63;
        Bs[cur][r * BS + c] = W[(size_t)(kt + r) * N + n0 + c];
      }
      __syncthreads();
    }
#endif

    v16bf a0 = load_a(&As[cur][0], wm, 0, AS);
    v16bf a1 = load_a(&As[cur][0], wm + 16, 0, AS);
    v16bf b0 = load_b(&Bs[cur][0], 0, wn, BS);
    v16bf b1 = load_b(&Bs[cur][0], 0, wn + 16, BS);
    acc[0][0] = WMMA_BF16(a0, b0, acc[0][0]);
    acc[0][1] = WMMA_BF16(a0, b1, acc[0][1]);
    acc[1][0] = WMMA_BF16(a1, b0, acc[1][0]);
    acc[1][1] = WMMA_BF16(a1, b1, acc[1][1]);
    __syncthreads();  // all reads of buf[cur] done before it is refilled
  }

#pragma unroll
  for (int mi = 0; mi < 2; ++mi)
#pragma unroll
    for (int ni = 0; ni < 2; ++ni) {
      int col = n0 + wn + ni * 16 + nlo;
      float bv = bias[col];
#pragma unroll
      for (int v = 0; v < 8; ++v) {
        int row = m0 + wm + mi * 16 + half * 8 + v;
        float val = acc[mi][ni][v] + bv;
        if (OUT_BF16)
          ((bf16*)Out)[(size_t)row * N + col] = (bf16)val;
        else
          ((float*)Out)[(size_t)row * N + col] = val;
      }
    }
}

// ---------------------------------------------------------------------------
// Flash attention (causal), one wave per (b, h, 32-row q tile).
// Q/KV bf16 in (B*T, 1024) row-major; score scale 1/sqrt(D) = 1/32.
// ---------------------------------------------------------------------------
__global__ __launch_bounds__(32) void flash_attn_kernel(
    const bf16* __restrict__ Qb, const bf16* __restrict__ KVb,
    bf16* __restrict__ Attn, int T) {
  constexpr int D = 1024, HDv = 64, KS = 72, PS = 40;
  __shared__ bf16 Qs[32 * KS];
  __shared__ bf16 Ks[2][32 * KS];
  __shared__ bf16 Ps[32 * PS];

  const int qBase = blockIdx.x * 32;
  const int h = blockIdx.y;
  const int b = blockIdx.z;
  const int lane = threadIdx.x;
  const int half = lane >> 4, nlo = lane & 15;

  const bf16* Qh = Qb + ((size_t)b * T) * D + h * HDv;
  const bf16* Kh = KVb + ((size_t)b * T) * D + h * HDv;
  const int jMax = qBase >> 5;

#if HAVE_TDM
  tdm_load_2d(lds_off(Qs), Qh + (size_t)qBase * D, 64, 32, (unsigned long long)D, 4, 3);
  tdm_load_2d(lds_off(&Ks[0][0]), Kh, 64, 32, (unsigned long long)D, 4, 3);
  TDM_WAIT(1);  // Q tile landed
#else
  for (int i = lane; i < 32 * 64; i += 32) {
    int r = i >> 6, c = i & 63;
    Qs[r * KS + c] = Qh[(size_t)(qBase + r) * D + c];
  }
  __syncthreads();
#endif

  v16bf aQ[2][2];
#pragma unroll
  for (int mi = 0; mi < 2; ++mi)
#pragma unroll
    for (int kk = 0; kk < 2; ++kk) aQ[mi][kk] = load_a(Qs, mi * 16, kk * 32, KS);

  float mrow[2][8], lrow[2][8];
  v8f O[2][4];
#pragma unroll
  for (int mi = 0; mi < 2; ++mi) {
#pragma unroll
    for (int v = 0; v < 8; ++v) { mrow[mi][v] = -3.0e38f; lrow[mi][v] = 0.0f; }
#pragma unroll
    for (int nd = 0; nd < 4; ++nd) O[mi][nd] = v8f{};
  }

  for (int j = 0; j <= jMax; ++j) {
    const int kBase = j * 32;
    const int cur = j & 1, nxt = cur ^ 1;
#if HAVE_TDM
    if (j + 1 <= jMax) {
      DS_FENCE();  // prior ds reads of Ks[nxt] complete before TDM refills it
      tdm_load_2d(lds_off(&Ks[nxt][0]), Kh + (size_t)(kBase + 32) * D, 64, 32,
                  (unsigned long long)D, 4, 3);
      TDM_WAIT(1);  // Ks[cur] landed
    } else {
      TDM_WAIT(0);
    }
#else
    __syncthreads();
    for (int i = lane; i < 32 * 64; i += 32) {
      int r = i >> 6, c = i & 63;
      Ks[cur][r * KS + c] = Kh[(size_t)(kBase + r) * D + c];
    }
    __syncthreads();
#endif

    // S = Q * KV^T (32x32): 8 WMMAs
    v8f s[2][2];
#pragma unroll
    for (int mi = 0; mi < 2; ++mi)
#pragma unroll
      for (int ni = 0; ni < 2; ++ni) s[mi][ni] = v8f{};
#pragma unroll
    for (int ni = 0; ni < 2; ++ni)
#pragma unroll
      for (int kk = 0; kk < 2; ++kk) {
        v16bf bK = load_bt(&Ks[cur][0], kk * 32, ni * 16, KS);
        s[0][ni] = WMMA_BF16(aQ[0][kk], bK, s[0][ni]);
        s[1][ni] = WMMA_BF16(aQ[1][kk], bK, s[1][ni]);
      }

    // scale + causal mask + online softmax
#pragma unroll
    for (int mi = 0; mi < 2; ++mi) {
#pragma unroll
      for (int v = 0; v < 8; ++v) {
        int q = qBase + mi * 16 + half * 8 + v;
        float s0 = s[mi][0][v] * 0.03125f;
        float s1 = s[mi][1][v] * 0.03125f;
        if (kBase + nlo > q) s0 = -1.0e30f;
        if (kBase + 16 + nlo > q) s1 = -1.0e30f;
        float rmax = fmaxf(s0, s1);
#pragma unroll
        for (int o = 8; o >= 1; o >>= 1) rmax = fmaxf(rmax, __shfl_xor(rmax, o, 32));
        float mnew = fmaxf(mrow[mi][v], rmax);
        float corr = __expf(mrow[mi][v] - mnew);
        mrow[mi][v] = mnew;
        float p0 = __expf(s0 - mnew);
        float p1 = __expf(s1 - mnew);
        float rs = p0 + p1;
#pragma unroll
        for (int o = 8; o >= 1; o >>= 1) rs += __shfl_xor(rs, o, 32);
        lrow[mi][v] = lrow[mi][v] * corr + rs;
#pragma unroll
        for (int nd = 0; nd < 4; ++nd) O[mi][nd][v] *= corr;
        int row = mi * 16 + half * 8 + v;
        Ps[row * PS + nlo] = (bf16)p0;
        Ps[row * PS + 16 + nlo] = (bf16)p1;
      }
    }
    __syncthreads();  // single-wave: just orders Ps stores vs loads

    // O += P * V (32x64): 8 WMMAs
#pragma unroll
    for (int nd = 0; nd < 4; ++nd) {
      v16bf bV = load_b(&Ks[cur][0], 0, nd * 16, KS);
#pragma unroll
      for (int mi = 0; mi < 2; ++mi) {
        v16bf aP = load_a(Ps, mi * 16, 0, PS);
        O[mi][nd] = WMMA_BF16(aP, bV, O[mi][nd]);
      }
    }
  }

  bf16* Ah = Attn + ((size_t)b * T) * D + h * HDv;
#pragma unroll
  for (int mi = 0; mi < 2; ++mi)
#pragma unroll
    for (int nd = 0; nd < 4; ++nd) {
      int d = nd * 16 + nlo;
#pragma unroll
      for (int v = 0; v < 8; ++v) {
        int q = qBase + mi * 16 + half * 8 + v;
        float val = O[mi][nd][v] / lrow[mi][v];
        Ah[(size_t)q * D + d] = (bf16)val;
      }
    }
}

// ---------------------------------------------------------------------------
extern "C" void kernel_launch(void* const* d_in, const int* in_sizes, int n_in,
                              void* d_out, int out_size, void* d_ws, size_t ws_size,
                              hipStream_t stream) {
  (void)in_sizes; (void)n_in; (void)out_size; (void)ws_size;
  const float* x  = (const float*)d_in[0];
  const float* Wq = (const float*)d_in[1];
  const float* bq = (const float*)d_in[2];
  const float* Wv = (const float*)d_in[3];
  const float* bv = (const float*)d_in[4];
  const float* Wo = (const float*)d_in[5];
  const float* bo = (const float*)d_in[6];
  float* out = (float*)d_out;

  const int B = 2, T = 2048, D = 1024, H = 16;
  const int M = B * T;                 // 4096
  const size_t MD = (size_t)M * D;     // 4M elems
  const size_t DD = (size_t)D * D;     // 1M elems

  bf16* xbf  = (bf16*)d_ws;            // [0,   4M)
  bf16* Wqb  = xbf + MD;               // [4M,  5M)
  bf16* Wvb  = Wqb + DD;               // [5M,  6M)
  bf16* Wob  = Wvb + DD;               // [6M,  7M)
  bf16* Qb   = Wob + DD;               // [7M, 11M)
  bf16* KVb  = Qb + MD;                // [11M,15M)
  bf16* Attn = KVb + MD;               // [15M,19M)

  cvt_bf16_kernel<<<(int)(MD / 1024), 256, 0, stream>>>(x, xbf, (int)MD);
  cvt_bf16_kernel<<<(int)(DD / 1024), 256, 0, stream>>>(Wq, Wqb, (int)DD);
  cvt_bf16_kernel<<<(int)(DD / 1024), 256, 0, stream>>>(Wv, Wvb, (int)DD);
  cvt_bf16_kernel<<<(int)(DD / 1024), 256, 0, stream>>>(Wo, Wob, (int)DD);

  dim3 gGemm(M / 64, D / 64);
  gemm_bias_kernel<true><<<gGemm, 128, 0, stream>>>(xbf, Wqb, bq, Qb, M, D, D);
  gemm_bias_kernel<true><<<gGemm, 128, 0, stream>>>(xbf, Wvb, bv, KVb, M, D, D);

  dim3 gAttn(T / 32, H, B);
  flash_attn_kernel<<<gAttn, 32, 0, stream>>>(Qb, KVb, Attn, T);

  gemm_bias_kernel<false><<<gGemm, 128, 0, stream>>>(Attn, Wob, bo, out, M, D, D);
}
